// ODERNN_66778151518455
// MI455X (gfx1250) — matrix operand order repaired and checked by reading, not compiled, not run on hardware
//
#include <hip/hip_runtime.h>
#include <hip/hip_bf16.h>

typedef __attribute__((ext_vector_type(16))) _Float16 v16h;
typedef __attribute__((ext_vector_type(8)))  _Float16 v8h;
typedef __attribute__((ext_vector_type(8)))  float    v8f;
typedef __attribute__((ext_vector_type(4)))  float    v4f;

#define B_TOT 4096
#define T_LEN 200
#define DS    32
#define DA    8
#define H     128
#define H3    384
#define LF    2
#define ALPHA_C 0.4f
#define BETA_C  0.8f

// fp16 weight workspace layout (offsets in halves)
#define OFF_FWR   0        // [LF][128][128]
#define OFF_FWZ   32768
#define OFF_FWH   65536
#define OFF_FWRL  98304    // [LF][128] last (time) row
#define OFF_FWZL  98560
#define OFF_FWHL  98816
#define OFF_WIH   99072    // [64][384] (K padded 40->64, transposed)
#define OFF_WHH   123648   // [128][384] (transposed)
#define OFF_DEC   172800   // [128][32]  (transposed)

#define OUTS_TOTAL (4096UL*200UL*32UL)

// ---------------- weight preparation (fp32 -> fp16, WMMA [K][N] layout) ----
__global__ void prep_flow_k(const float* __restrict__ Wr, const float* __restrict__ Wz,
                            const float* __restrict__ Wh, _Float16* __restrict__ w) {
  int i = blockIdx.x * 256 + threadIdx.x;
  const int tot = LF * (H + 1) * H;
  if (i >= tot) return;
  int l = i / ((H + 1) * H);
  int rem = i - l * (H + 1) * H;
  int k = rem / H, n = rem - k * H;
  _Float16 r = (_Float16)Wr[i], z = (_Float16)Wz[i], h = (_Float16)Wh[i];
  if (k < H) {
    int o = l * H * H + k * H + n;
    w[OFF_FWR + o] = r; w[OFF_FWZ + o] = z; w[OFF_FWH + o] = h;
  } else {
    int o = l * H + n;
    w[OFF_FWRL + o] = r; w[OFF_FWZL + o] = z; w[OFF_FWHL + o] = h;
  }
}

__global__ void prep_gru_k(const float* __restrict__ Wih, const float* __restrict__ Whh,
                           const float* __restrict__ decW, _Float16* __restrict__ w) {
  int i = blockIdx.x * 256 + threadIdx.x;
  const int N1 = 64 * H3, N2 = H * H3, N3 = H * DS;
  if (i < N1) {                       // Wih^T padded:  [64][384]
    int k = i / H3, n = i - k * H3;
    w[OFF_WIH + i] = (k < (DS + DA)) ? (_Float16)Wih[n * (DS + DA) + k] : (_Float16)0.0f;
  } else if (i < N1 + N2) {           // Whh^T: [128][384]
    int j = i - N1; int k = j / H3, n = j - k * H3;
    w[OFF_WHH + j] = (_Float16)Whh[n * H + k];
  } else if (i < N1 + N2 + N3) {      // dec_W^T: [128][32]
    int j = i - N1 - N2; int k = j / DS, n = j - k * DS;
    w[OFF_DEC + j] = (_Float16)decW[n * H + k];
  }
}

// ---------------- WMMA helpers -------------------------------------------
// A fragment (16x32 f16) from LDS tile with row stride `stride` halves.
// ISA: lanes 0-15 -> M=lane, K=k0+0..7 then k0+16..23;
//      lanes 16-31 -> M=lane-16, K=k0+8..15 then k0+24..31.
__device__ __forceinline__ v16h frag_A(const _Float16* base, int stride, int lane, int k0) {
  const int mm = lane & 15;
  const int ho = (lane >> 4) << 3;
  const _Float16* p = base + mm * stride + k0 + ho;
  v8h lo = *(const v8h*)p;
  v8h hv = *(const v8h*)(p + 16);
  v16h r;
#pragma unroll
  for (int i = 0; i < 8; i++) { r[i] = lo[i]; r[i + 8] = hv[i]; }
  return r;
}

// B fragment (32x16 f16): lane l holds column N=n0+(l&15), K=k0+ho+0..7 then k0+16+ho+0..7 (ho=8*(l/16)).
__device__ __forceinline__ v16h frag_B(const _Float16* W, int ldn, int lane, int k0, int n0) {
  const int nn = lane & 15;
  const int ho = (lane >> 4) << 3;
  const _Float16* p = W + (size_t)(k0 + ho) * ldn + n0 + nn;
  v16h r;
#pragma unroll
  for (int i = 0; i < 8; i++) { r[i] = p[(size_t)i * ldn]; r[i + 8] = p[(size_t)(i + 16) * ldn]; }
  return r;
}

__device__ __forceinline__ v8f wmma_f16(v16h a, v16h b, v8f c) {
  return __builtin_amdgcn_wmma_f32_16x16x32_f16(false, a, false, b, (short)0, c, false, false);
}

// gfx1250 has a hardware v_tanh_f32 trans op
__device__ __forceinline__ float fast_tanhf(float x) {
#if __has_builtin(__builtin_amdgcn_tanhf)
  return __builtin_amdgcn_tanhf(x);
#else
  return tanhf(x);
#endif
}
__device__ __forceinline__ float sigf(float x) { return 0.5f + 0.5f * fast_tanhf(0.5f * x); }

// ---------------- main scan kernel ---------------------------------------
// 2 waves / WG (64 threads); each wave owns 16 batch rows; state kept in LDS:
//   hf   : fp32 authoritative h         (8KB)
//   h16  : fp16 mirror (A operand)      (4KB)
//   rh16 : r*h | GRU r | h_new staging  (4KB)
//   gz16 : z staging                    (4KB)
//   x16  : concat(s,a) padded 40->64    (2KB)
__global__ __launch_bounds__(64) void odernn_scan(
    const _Float16* __restrict__ ws,
    const float* __restrict__ s,  const float* __restrict__ a,  const float* __restrict__ t,
    const float* __restrict__ fbr, const float* __restrict__ fbz, const float* __restrict__ fbh,
    const float* __restrict__ ftw,
    const float* __restrict__ bih, const float* __restrict__ bhh,
    const float* __restrict__ decb,
    float* __restrict__ out)
{
  __shared__ __attribute__((aligned(16))) float    sh_hf [2][16 * H];
  __shared__ __attribute__((aligned(16))) _Float16 sh_h16[2][16 * H];
  __shared__ __attribute__((aligned(16))) _Float16 sh_rh [2][16 * H];
  __shared__ __attribute__((aligned(16))) _Float16 sh_gz [2][16 * H];
  __shared__ __attribute__((aligned(16))) _Float16 sh_x  [2][16 * 64];
  __shared__ float sh_tt[2][16];

  const int lane = threadIdx.x & 31;
  const int wave = threadIdx.x >> 5;
  const int b0   = (blockIdx.x * 2 + wave) * 16;

  float*    hf   = sh_hf[wave];
  _Float16* h16  = sh_h16[wave];
  _Float16* rh16 = sh_rh[wave];
  _Float16* gz16 = sh_gz[wave];
  _Float16* x16  = sh_x[wave];
  float*    ttl  = sh_tt[wave];

  const int col = lane & 15;                // N within tile (C layout)
  const int hi8 = (lane >> 4) << 3;         // 0 or 8 (C layout row offset)

  // ---- init: h = 0
  for (int idx = lane; idx < 16 * H;  idx += 32) { hf[idx] = 0.0f; h16[idx] = (_Float16)0.0f; }
  for (int idx = lane; idx < 16 * 64; idx += 32) x16[idx] = (_Float16)0.0f;

#pragma unroll 1
  for (int ti = 0; ti < T_LEN; ++ti) {
    // ---- stage per-step inputs
    if (lane < 16) ttl[lane] = t[(size_t)(b0 + lane) * T_LEN + ti];
    {
      int row = lane & 15;
      if (lane < 16) {
        const float* sp = s + ((size_t)(b0 + row) * T_LEN + ti) * DS;
#pragma unroll
        for (int c = 0; c < DS; c++) x16[row * 64 + c] = (_Float16)sp[c];
      } else {
        const float* ap = a + ((size_t)(b0 + row) * T_LEN + ti) * DA;
#pragma unroll
        for (int c = 0; c < DA; c++) x16[row * 64 + 32 + c] = (_Float16)ap[c];
      }
    }
    if (ti + 1 < T_LEN && lane < 16) {      // global_prefetch_b8 for next step
      __builtin_prefetch(s + ((size_t)(b0 + lane) * T_LEN + ti + 1) * DS, 0, 0);
      __builtin_prefetch(a + ((size_t)(b0 + lane) * T_LEN + ti + 1) * DA, 0, 0);
    }

    float ttv[8];
#pragma unroll
    for (int j = 0; j < 8; j++) ttv[j] = ttl[j + hi8];

    // =================== flow blocks ===================
#pragma unroll 1
    for (int l = 0; l < LF; l++) {
      const _Float16* Wr = ws + OFF_FWR + l * H * H;
      const _Float16* Wz = ws + OFF_FWZ + l * H * H;
      const _Float16* Wh = ws + OFF_FWH + l * H * H;

      v16h ah[4];
#pragma unroll
      for (int k = 0; k < 4; k++) ah[k] = frag_A(h16, H, lane, k * 32);

      // r = BETA*sigmoid(h@Wr + tt*WrLast + br);  rh16 = r*h
#pragma unroll 1
      for (int nt = 0; nt < 8; nt++) {
        v8f acc = {};
#pragma unroll
        for (int k = 0; k < 4; k++) acc = wmma_f16(ah[k], frag_B(Wr, H, lane, k * 32, nt * 16), acc);
        float wrl = (float)ws[OFF_FWRL + l * H + nt * 16 + col];
        float brv = fbr[l * H + nt * 16 + col];
#pragma unroll
        for (int j = 0; j < 8; j++) {
          int o = (j + hi8) * H + nt * 16 + col;
          float rv = BETA_C * sigf(acc[j] + ttv[j] * wrl + brv);
          rh16[o] = (_Float16)(rv * hf[o]);
        }
      }
      // z = ALPHA*sigmoid(h@Wz + tt*WzLast + bz) -> gz16
#pragma unroll 1
      for (int nt = 0; nt < 8; nt++) {
        v8f acc = {};
#pragma unroll
        for (int k = 0; k < 4; k++) acc = wmma_f16(ah[k], frag_B(Wz, H, lane, k * 32, nt * 16), acc);
        float wzl = (float)ws[OFF_FWZL + l * H + nt * 16 + col];
        float bzv = fbz[l * H + nt * 16 + col];
#pragma unroll
        for (int j = 0; j < 8; j++) {
          int o = (j + hi8) * H + nt * 16 + col;
          gz16[o] = (_Float16)(ALPHA_C * sigf(acc[j] + ttv[j] * wzl + bzv));
        }
      }
      // u = tanh(rh@Wh + tt*WhLast + bh);  h += tanh(tw*tt)*z*(u-h)
      v16h au[4];
#pragma unroll
      for (int k = 0; k < 4; k++) au[k] = frag_A(rh16, H, lane, k * 32);
#pragma unroll 1
      for (int nt = 0; nt < 8; nt++) {
        v8f acc = {};
#pragma unroll
        for (int k = 0; k < 4; k++) acc = wmma_f16(au[k], frag_B(Wh, H, lane, k * 32, nt * 16), acc);
        float whl = (float)ws[OFF_FWHL + l * H + nt * 16 + col];
        float bhv = fbh[l * H + nt * 16 + col];
        float tw  = ftw[l * H + nt * 16 + col];
#pragma unroll
        for (int j = 0; j < 8; j++) {
          int o = (j + hi8) * H + nt * 16 + col;
          float uv  = fast_tanhf(acc[j] + ttv[j] * whl + bhv);
          float phi = fast_tanhf(tw * ttv[j]);
          float ho  = hf[o];
          float hn  = ho + phi * (float)gz16[o] * (uv - ho);
          hf[o] = hn;
          h16[o] = (_Float16)hn;
        }
      }
    }

    // ---- hids output (post-flow h), vectorized b128 copies
    {
      int m  = lane & 15;
      int c0 = (lane >> 4) * 64;
      float* dst = out + OUTS_TOTAL + ((size_t)(b0 + m) * T_LEN + ti) * H + c0;
      const float* src = hf + m * H + c0;
#pragma unroll
      for (int c = 0; c < 64; c += 4) *(v4f*)(dst + c) = *(const v4f*)(src + c);
    }

    // =================== GRU cell ===================
    {
      v16h ax[2];
#pragma unroll
      for (int k = 0; k < 2; k++) ax[k] = frag_A(x16, 64, lane, k * 32);
      v16h ah[4];
#pragma unroll
      for (int k = 0; k < 4; k++) ah[k] = frag_A(h16, H, lane, k * 32);

      // r (tiles 0..7 -> rh16) and z (tiles 8..15 -> gz16)
#pragma unroll 1
      for (int nt = 0; nt < 16; nt++) {
        v8f gi = {};
#pragma unroll
        for (int k = 0; k < 2; k++) gi = wmma_f16(ax[k], frag_B(ws + OFF_WIH, H3, lane, k * 32, nt * 16), gi);
        v8f gh = {};
#pragma unroll
        for (int k = 0; k < 4; k++) gh = wmma_f16(ah[k], frag_B(ws + OFF_WHH, H3, lane, k * 32, nt * 16), gh);
        float biv = bih[nt * 16 + col];
        float bhv = bhh[nt * 16 + col];
        _Float16* dst = (nt < 8) ? rh16 : gz16;
        int cb = (nt & 7) * 16 + col;
#pragma unroll
        for (int j = 0; j < 8; j++)
          dst[(j + hi8) * H + cb] = (_Float16)sigf(gi[j] + gh[j] + biv + bhv);
      }
      // n-tiles 16..23: h_new = (1-z)*n + z*h ; stage h_new fp16 into rh16
#pragma unroll 1
      for (int q = 0; q < 8; q++) {
        const int nt = 16 + q;
        v8f gi = {};
#pragma unroll
        for (int k = 0; k < 2; k++) gi = wmma_f16(ax[k], frag_B(ws + OFF_WIH, H3, lane, k * 32, nt * 16), gi);
        v8f gh = {};
#pragma unroll
        for (int k = 0; k < 4; k++) gh = wmma_f16(ah[k], frag_B(ws + OFF_WHH, H3, lane, k * 32, nt * 16), gh);
        float biv = bih[nt * 16 + col];
        float bhv = bhh[nt * 16 + col];
#pragma unroll
        for (int j = 0; j < 8; j++) {
          int o = (j + hi8) * H + q * 16 + col;
          float rv = (float)rh16[o];
          float zv = (float)gz16[o];
          float nn = fast_tanhf(gi[j] + biv + rv * (gh[j] + bhv));
          float ho = hf[o];
          float hn = (1.0f - zv) * nn + zv * ho;
          hf[o]   = hn;
          rh16[o] = (_Float16)hn;      // staging: h16 must stay old until all matmuls done
        }
      }
      // commit: h16 <- rh16 (vector copies)
#pragma unroll 1
      for (int idx = lane * 8; idx < 16 * H; idx += 32 * 8)
        *(v8h*)(h16 + idx) = *(const v8h*)(rh16 + idx);
    }

    // =================== decoder ===================
    {
      v16h ah[4];
#pragma unroll
      for (int k = 0; k < 4; k++) ah[k] = frag_A(h16, H, lane, k * 32);
#pragma unroll 1
      for (int nt = 0; nt < 2; nt++) {
        v8f acc = {};
#pragma unroll
        for (int k = 0; k < 4; k++) acc = wmma_f16(ah[k], frag_B(ws + OFF_DEC, DS, lane, k * 32, nt * 16), acc);
        float db = decb[nt * 16 + col];
#pragma unroll
        for (int j = 0; j < 8; j++)
          out[((size_t)(b0 + j + hi8) * T_LEN + ti) * DS + nt * 16 + col] = acc[j] + db;
      }
    }
  }
}

// ---------------- launch ---------------------------------------------------
extern "C" void kernel_launch(void* const* d_in, const int* in_sizes, int n_in,
                              void* d_out, int out_size, void* d_ws, size_t ws_size,
                              hipStream_t stream) {
  const float* s    = (const float*)d_in[0];
  const float* a    = (const float*)d_in[1];
  const float* t    = (const float*)d_in[2];
  const float* fWr  = (const float*)d_in[3];
  const float* fbr  = (const float*)d_in[4];
  const float* fWz  = (const float*)d_in[5];
  const float* fbz  = (const float*)d_in[6];
  const float* fWh  = (const float*)d_in[7];
  const float* fbh  = (const float*)d_in[8];
  const float* ftw  = (const float*)d_in[9];
  const float* Wih  = (const float*)d_in[10];
  const float* Whh  = (const float*)d_in[11];
  const float* bih  = (const float*)d_in[12];
  const float* bhh  = (const float*)d_in[13];
  const float* decW = (const float*)d_in[14];
  const float* decb = (const float*)d_in[15];
  float* out = (float*)d_out;
  _Float16* w16 = (_Float16*)d_ws;

  prep_flow_k<<<(LF * (H + 1) * H + 255) / 256, 256, 0, stream>>>(fWr, fWz, fWh, w16);
  prep_gru_k<<<(64 * H3 + H * H3 + H * DS + 255) / 256, 256, 0, stream>>>(Wih, Whh, decW, w16);

  // 4096 rows / (2 waves * 16 rows) = 128 workgroups of 64 threads
  odernn_scan<<<128, 64, 0, stream>>>(w16, s, a, t, fbr, fbz, fbh, ftw, bih, bhh, decb, out);
}
